// ComplexCrossAttention_3539053051945
// MI455X (gfx1250) — matrix operand order repaired and checked by reading, not compiled, not run on hardware
//
#include <hip/hip_runtime.h>

// ---------------- problem constants ----------------
constexpr int B_  = 2;
constexpr int S_  = 2048;
constexpr int LC_ = 1024;
constexpr int F_  = 1024;
constexpr int DC_ = 768;
constexpr int H_  = 16;
constexpr int HD_ = 64;

typedef __attribute__((ext_vector_type(16))) __bf16 v16bf;
typedef __attribute__((ext_vector_type(8)))  float  v8f;

union Frag { uint4 u[2]; v16bf v; };

// hardware bf16 convert (RNE) -> lets backend use v_cvt_*bf16_f32
__device__ __forceinline__ unsigned short f2bf(float f) {
    union { __bf16 b; unsigned short s; } v;
    v.b = (__bf16)f;
    return v.s;
}

__device__ __forceinline__ uint4 negbf8(uint4 a) {
    a.x ^= 0x80008000u; a.y ^= 0x80008000u;
    a.z ^= 0x80008000u; a.w ^= 0x80008000u;
    return a;
}

__device__ __forceinline__ v8f wmma_bf16(const Frag& a, const Frag& b, v8f c) {
    return __builtin_amdgcn_wmma_f32_16x16x32_bf16(false, a.v, false, b.v,
                                                   (short)0, c, false, false);
}

// A-matrix fragment (16x32 bf16): lanes 0-15 hold K {0..7,16..23}, lanes 16-31 K {8..15,24..31}
__device__ __forceinline__ Frag load_fragA(const unsigned short* base, int row, int lda,
                                           int ks, int lane) {
    Frag f;
    const int k0 = ks + ((lane < 16) ? 0 : 8);
    const unsigned short* p = base + row * lda + k0;
    f.u[0] = *(const uint4*)(p);
    f.u[1] = *(const uint4*)(p + 16);
    return f;
}

// B-matrix fragment (32x16 bf16) from Bt layout [n][k]: lanes 0-15 K {0..7,16..23}, lanes 16-31 K {8..15,24..31}
__device__ __forceinline__ Frag load_fragB(const unsigned short* base, int col, int ldb,
                                           int ks, int lane) {
    Frag f;
    const int k0 = ks + ((lane < 16) ? 0 : 8);
    const unsigned short* p = base + col * ldb + k0;
    f.u[0] = *(const uint4*)(p);
    f.u[1] = *(const uint4*)(p + 16);
    return f;
}

// ---------------- Tensor Data Mover path (guarded) ----------------
#if __has_builtin(__builtin_amdgcn_tensor_load_to_lds)
#define USE_TDM 1
typedef unsigned int v4u __attribute__((ext_vector_type(4)));
typedef int          v4i __attribute__((ext_vector_type(4)));
typedef int          v8i __attribute__((ext_vector_type(8)));

// 64x64 bf16 tile, global row stride `row_stride_elems` (elements), into LDS with
// 16B padding after each 128B row (LDS row stride 72 halfs = 144B).
__device__ __forceinline__ void tdm_tile_64x64_bf16(const unsigned short* gsrc,
                                                    unsigned short* ldst,
                                                    unsigned row_stride_elems) {
    const unsigned long long ga = (unsigned long long)(const void*)gsrc;
    const unsigned lds = (unsigned)(unsigned long long)(void*)ldst;
    v4u g0;
    g0[0] = 1u;                                              // count=1 (valid user D#)
    g0[1] = lds;                                             // lds_addr
    g0[2] = (unsigned)(ga & 0xffffffffu);                    // global_addr[31:0]
    g0[3] = (unsigned)((ga >> 32) & 0x01ffffffu)             // global_addr[56:32]
          | (2u << 30);                                      // type = 2 ("image")
    v8i g1;
    g1[0] = (int)((1u << 16)      // data_size = 2 bytes
                | (1u << 20)      // pad_enable
                | (4u << 22)      // pad_interval: every 32 DWORDs (=128B row)
                | (3u << 25));    // pad_amount: 4 DWORDs (=16B)
    g1[1] = (int)(64u << 16);     // tensor_dim0[15:0] (=64) at bits 63:48
    g1[2] = (int)(64u << 16);     // tensor_dim1[15:0] (=64) at bits 95:80
    g1[3] = (int)(64u << 16);     // tile_dim0 = 64 at bits 127:112
    g1[4] = (int)(64u);           // tile_dim1 = 64 (tile_dim2 = 0)
    g1[5] = (int)row_stride_elems;// tensor_dim0_stride[31:0]
    g1[6] = 0;
    g1[7] = 0;
    const v4i z4 = {0, 0, 0, 0};
    const v8i z8 = {0, 0, 0, 0, 0, 0, 0, 0};
    // 6-arg (clang-23 / therock) form: (g0, g1, g2, g3, g4, cpol)
    __builtin_amdgcn_tensor_load_to_lds(g0, g1, z4, z4, z8, 0);
}
#else
#define USE_TDM 0
#endif

// =====================================================================
// Complex GEMM:  Dr = Ar@Wr - Ai@Wi + br ;  Di = Ar@Wi + Ai@Wr + bi
// Block tile 64x64, BK=32, 8 waves: wave -> (mt = w>>1, n-tiles {2*(w&1), 2*(w&1)+1})
// OUT_MODE: 0 = bf16 row-major (M x N) ; 1 = bf16 per-head transposed (B,H,HD,LC)
//           2 = interleaved complex float32 to d_out
// =====================================================================
template<int OUT_MODE, bool A_BF16>
__global__ __launch_bounds__(256) void cgemm_kernel(
    const void* __restrict__ Ar_, const void* __restrict__ Ai_,
    const float* __restrict__ Wr, const float* __restrict__ Wi,
    const float* __restrict__ br, const float* __restrict__ bi,
    void* __restrict__ Cr_, void* __restrict__ Ci_,
    int M, int K, int N)
{
    __shared__ unsigned short Asr[64 * 40];
    __shared__ unsigned short Asi[64 * 40];
    __shared__ unsigned short Wtr[64 * 40];
    __shared__ unsigned short Wti[64 * 40];

    const int tid  = threadIdx.x;
    const int lane = tid & 31;
    const int ln   = lane & 15;
    const int wave = tid >> 5;
    const int mt   = wave >> 1;
    const int ntp  = (wave & 1) * 2;
    const int m0   = blockIdx.y * 64;
    const int n0   = blockIdx.x * 64;

    v8f accR0 = {}, accR1 = {}, accI0 = {}, accI1 = {};

    for (int k0 = 0; k0 < K; k0 += 32) {
        // ---- A tile (64 rows x 32 k) -> bf16 LDS, row stride 40 ----
        {
            const int r = tid >> 2;
            const int c = (tid & 3) * 8;
            const size_t src = (size_t)(m0 + r) * K + k0 + c;
            if constexpr (A_BF16) {
                const unsigned short* Ar = (const unsigned short*)Ar_;
                const unsigned short* Ai = (const unsigned short*)Ai_;
                *(uint4*)&Asr[r * 40 + c] = *(const uint4*)&Ar[src];
                *(uint4*)&Asi[r * 40 + c] = *(const uint4*)&Ai[src];
                if (k0 + 32 < K) {                       // prefetch next K tile
                    __builtin_prefetch(&Ar[src + 32], 0, 1);
                    __builtin_prefetch(&Ai[src + 32], 0, 1);
                }
            } else {
                const float* Ar = (const float*)Ar_;
                const float* Ai = (const float*)Ai_;
                float4 a0 = *(const float4*)&Ar[src];
                float4 a1 = *(const float4*)&Ar[src + 4];
                float4 b0 = *(const float4*)&Ai[src];
                float4 b1 = *(const float4*)&Ai[src + 4];
                union { unsigned short s[8]; uint4 q; } pr, pi;
                pr.s[0] = f2bf(a0.x); pr.s[1] = f2bf(a0.y); pr.s[2] = f2bf(a0.z); pr.s[3] = f2bf(a0.w);
                pr.s[4] = f2bf(a1.x); pr.s[5] = f2bf(a1.y); pr.s[6] = f2bf(a1.z); pr.s[7] = f2bf(a1.w);
                pi.s[0] = f2bf(b0.x); pi.s[1] = f2bf(b0.y); pi.s[2] = f2bf(b0.z); pi.s[3] = f2bf(b0.w);
                pi.s[4] = f2bf(b1.x); pi.s[5] = f2bf(b1.y); pi.s[6] = f2bf(b1.z); pi.s[7] = f2bf(b1.w);
                *(uint4*)&Asr[r * 40 + c] = pr.q;
                *(uint4*)&Asi[r * 40 + c] = pi.q;
                if (k0 + 32 < K) {
                    __builtin_prefetch(&Ar[src + 32], 0, 1);
                    __builtin_prefetch(&Ai[src + 32], 0, 1);
                }
            }
        }
        // ---- W tile (32 k x 64 n), transposed into LDS [n][k], stride 40 ----
        {
            const int k  = tid >> 3;
            const int nn = (tid & 7) * 8;
            const size_t src = (size_t)(k0 + k) * N + n0 + nn;
            float4 w0 = *(const float4*)&Wr[src];
            float4 w1 = *(const float4*)&Wr[src + 4];
            float4 v0 = *(const float4*)&Wi[src];
            float4 v1 = *(const float4*)&Wi[src + 4];
            Wtr[(nn + 0) * 40 + k] = f2bf(w0.x); Wtr[(nn + 1) * 40 + k] = f2bf(w0.y);
            Wtr[(nn + 2) * 40 + k] = f2bf(w0.z); Wtr[(nn + 3) * 40 + k] = f2bf(w0.w);
            Wtr[(nn + 4) * 40 + k] = f2bf(w1.x); Wtr[(nn + 5) * 40 + k] = f2bf(w1.y);
            Wtr[(nn + 6) * 40 + k] = f2bf(w1.z); Wtr[(nn + 7) * 40 + k] = f2bf(w1.w);
            Wti[(nn + 0) * 40 + k] = f2bf(v0.x); Wti[(nn + 1) * 40 + k] = f2bf(v0.y);
            Wti[(nn + 2) * 40 + k] = f2bf(v0.z); Wti[(nn + 3) * 40 + k] = f2bf(v0.w);
            Wti[(nn + 4) * 40 + k] = f2bf(v1.x); Wti[(nn + 5) * 40 + k] = f2bf(v1.y);
            Wti[(nn + 6) * 40 + k] = f2bf(v1.z); Wti[(nn + 7) * 40 + k] = f2bf(v1.w);
            if (k0 + 32 < K) {
                __builtin_prefetch(&Wr[src + (size_t)32 * N], 0, 1);
                __builtin_prefetch(&Wi[src + (size_t)32 * N], 0, 1);
            }
        }
        __syncthreads();

        Frag ar = load_fragA(Asr, mt * 16 + ln, 40, 0, lane);
        Frag ai = load_fragA(Asi, mt * 16 + ln, 40, 0, lane);
        Frag am; am.u[0] = negbf8(ai.u[0]); am.u[1] = negbf8(ai.u[1]);

        Frag wr0 = load_fragB(Wtr, (ntp + 0) * 16 + ln, 40, 0, lane);
        Frag wi0 = load_fragB(Wti, (ntp + 0) * 16 + ln, 40, 0, lane);
        Frag wr1 = load_fragB(Wtr, (ntp + 1) * 16 + ln, 40, 0, lane);
        Frag wi1 = load_fragB(Wti, (ntp + 1) * 16 + ln, 40, 0, lane);

        accR0 = wmma_bf16(ar, wr0, accR0);
        accR0 = wmma_bf16(am, wi0, accR0);
        accI0 = wmma_bf16(ar, wi0, accI0);
        accI0 = wmma_bf16(ai, wr0, accI0);
        accR1 = wmma_bf16(ar, wr1, accR1);
        accR1 = wmma_bf16(am, wi1, accR1);
        accI1 = wmma_bf16(ar, wi1, accI1);
        accI1 = wmma_bf16(ai, wr1, accI1);
        __syncthreads();
    }

    // ---------------- epilogue ----------------
    const int hi8 = (lane >= 16) ? 8 : 0;
#pragma unroll
    for (int j = 0; j < 2; ++j) {
        const v8f aR = j ? accR1 : accR0;
        const v8f aI = j ? accI1 : accI0;
        const int col = n0 + (ntp + j) * 16 + ln;
        const float fbr = br[col];
        const float fbi = bi[col];
#pragma unroll
        for (int r = 0; r < 8; ++r) {
            const int row = m0 + mt * 16 + r + hi8;
            const float vr = aR[r] + fbr;
            const float vi = aI[r] + fbi;
            if constexpr (OUT_MODE == 0) {
                ((unsigned short*)Cr_)[(size_t)row * N + col] = f2bf(vr);
                ((unsigned short*)Ci_)[(size_t)row * N + col] = f2bf(vi);
            } else if constexpr (OUT_MODE == 1) {
                const int bb = row / LC_, l = row % LC_;
                const int hh = col >> 6, dd = col & 63;
                const size_t dst = (((size_t)bb * H_ + hh) * HD_ + dd) * LC_ + l;
                ((unsigned short*)Cr_)[dst] = f2bf(vr);
                ((unsigned short*)Ci_)[dst] = f2bf(vi);
            } else {
                float* out = (float*)Cr_;
                const size_t idx = (size_t)row * N + col;
                out[2 * idx]     = vr;
                out[2 * idx + 1] = vi;
            }
        }
    }
}

// =====================================================================
// Fused flash attention: one block = (b,h) x 64-query chunk, 8 waves.
// scores = (Qr.Kr^T + Qi.Ki^T)/8 + mask ; online softmax ; out = P@Vr, P@Vi
// =====================================================================
constexpr size_t ATTN_SMEM = (size_t)64 * 68 * 4   // Ss (f32 scores)
                           + 3 * 64 * 4            // mrow/srow/corr
                           + (size_t)7 * 64 * 72 * 2; // Qsr,Qsi,Ksr,Ksi,Vsr,Vsi,Ps

__global__ __launch_bounds__(256) void attn_kernel(
    const unsigned short* __restrict__ Qr, const unsigned short* __restrict__ Qi,
    const unsigned short* __restrict__ Kr, const unsigned short* __restrict__ Ki,
    const unsigned short* __restrict__ Vtr, const unsigned short* __restrict__ Vti,
    const float* __restrict__ mask,
    unsigned short* __restrict__ Or, unsigned short* __restrict__ Oi)
{
    extern __shared__ __align__(16) char smem[];
    float* Ss   = (float*)smem;                           // [64][68]
    float* mrow = (float*)(smem + 64 * 68 * 4);
    float* srow = mrow + 64;
    float* corr = srow + 64;
    unsigned short* Qsr = (unsigned short*)(smem + 64 * 68 * 4 + 3 * 64 * 4);
    unsigned short* Qsi = Qsr + 64 * 72;
    unsigned short* Ksr = Qsi + 64 * 72;
    unsigned short* Ksi = Ksr + 64 * 72;
    unsigned short* Vsr = Ksi + 64 * 72;
    unsigned short* Vsi = Vsr + 64 * 72;
    unsigned short* Ps  = Vsi + 64 * 72;

    const int tid  = threadIdx.x;
    const int lane = tid & 31;
    const int ln   = lane & 15;
    const int wave = tid >> 5;
    const int mt   = wave >> 1;
    const int ntp  = (wave & 1) * 2;
    const int hi8  = (lane >= 16) ? 8 : 0;
    const int arow = mt * 16 + ln;

    const int q0 = blockIdx.x * 64;
    const int bh = blockIdx.y;
    const int b  = bh >> 4;   // H_ = 16
    const int h  = bh & 15;

    // ---- load Q tiles (bf16, 64x64) ----
#if USE_TDM
    if (wave == 0) {
        const size_t qsrc = (size_t)(b * S_ + q0) * F_ + h * 64;
        tdm_tile_64x64_bf16(&Qr[qsrc], Qsr, (unsigned)F_);
        tdm_tile_64x64_bf16(&Qi[qsrc], Qsi, (unsigned)F_);
    }
#else
    for (int i = tid; i < 512; i += 256) {
        const int r = i >> 3;
        const int c = (i & 7) * 8;
        const size_t src = (size_t)(b * S_ + q0 + r) * F_ + h * 64 + c;
        *(uint4*)&Qsr[r * 72 + c] = *(const uint4*)&Qr[src];
        *(uint4*)&Qsi[r * 72 + c] = *(const uint4*)&Qi[src];
    }
#endif
    if (tid < 64) { mrow[tid] = -1e30f; srow[tid] = 0.0f; }

    v8f oR0 = {}, oR1 = {}, oI0 = {}, oI1 = {};

    for (int l0 = 0; l0 < LC_; l0 += 64) {
        // ---- load K and V^T tiles ----
#if USE_TDM
        if (wave == 0) {
            const size_t ksrc = (size_t)(b * LC_ + l0) * F_ + h * 64;
            const size_t vsrc = (size_t)bh * HD_ * LC_ + l0;
            tdm_tile_64x64_bf16(&Kr[ksrc],  Ksr, (unsigned)F_);
            tdm_tile_64x64_bf16(&Ki[ksrc],  Ksi, (unsigned)F_);
            tdm_tile_64x64_bf16(&Vtr[vsrc], Vsr, (unsigned)LC_);
            tdm_tile_64x64_bf16(&Vti[vsrc], Vsi, (unsigned)LC_);
            __builtin_amdgcn_s_wait_tensorcnt(0);   // also covers Q tiles on first iter
        }
        __syncthreads();
#else
        for (int i = tid; i < 512; i += 256) {
            const int r = i >> 3;
            const int c = (i & 7) * 8;
            const size_t ksrc = (size_t)(b * LC_ + l0 + r) * F_ + h * 64 + c;
            *(uint4*)&Ksr[r * 72 + c] = *(const uint4*)&Kr[ksrc];
            *(uint4*)&Ksi[r * 72 + c] = *(const uint4*)&Ki[ksrc];
            const size_t vsrc = ((size_t)bh * HD_ + r) * LC_ + l0 + c;
            *(uint4*)&Vsr[r * 72 + c] = *(const uint4*)&Vtr[vsrc];
            *(uint4*)&Vsi[r * 72 + c] = *(const uint4*)&Vti[vsrc];
        }
        __syncthreads();
#endif

        // ---- scores: (Qr.Kr^T + Qi.Ki^T) ----
        v8f sR0 = {}, sR1 = {};
#pragma unroll
        for (int ks = 0; ks < 64; ks += 32) {
            Frag qr  = load_fragA(Qsr, arow, 72, ks, lane);
            Frag qi  = load_fragA(Qsi, arow, 72, ks, lane);
            Frag kr0 = load_fragB(Ksr, (ntp + 0) * 16 + ln, 72, ks, lane);
            Frag ki0 = load_fragB(Ksi, (ntp + 0) * 16 + ln, 72, ks, lane);
            Frag kr1 = load_fragB(Ksr, (ntp + 1) * 16 + ln, 72, ks, lane);
            Frag ki1 = load_fragB(Ksi, (ntp + 1) * 16 + ln, 72, ks, lane);
            sR0 = wmma_bf16(qr, kr0, sR0);
            sR0 = wmma_bf16(qi, ki0, sR0);
            sR1 = wmma_bf16(qr, kr1, sR1);
            sR1 = wmma_bf16(qi, ki1, sR1);
        }
        // scale + additive mask, spill score tile to LDS (f32)
#pragma unroll
        for (int j = 0; j < 2; ++j) {
            const v8f s = j ? sR1 : sR0;
            const int col = (ntp + j) * 16 + ln;
            const float mcol = mask[b * LC_ + l0 + col];
            const float madd = (1.0f - mcol) * -1e9f;
#pragma unroll
            for (int r = 0; r < 8; ++r) {
                const int row = mt * 16 + r + hi8;
                Ss[row * 68 + col] = s[r] * 0.125f + madd;
            }
        }
        __syncthreads();

        // ---- online softmax: one thread per query row ----
        if (tid < 64) {
            const int row = tid;
            const float mold = mrow[row];
            float tmax = -1e30f;
#pragma unroll 8
            for (int c = 0; c < 64; ++c) tmax = fmaxf(tmax, Ss[row * 68 + c]);
            const float mnew = fmaxf(mold, tmax);
            const float cf   = __expf(mold - mnew);
            float s = srow[row] * cf;
#pragma unroll 8
            for (int c = 0; c < 64; ++c) {
                const float p = __expf(Ss[row * 68 + c] - mnew);
                s += p;
                Ps[row * 72 + c] = f2bf(p);
            }
            mrow[row] = mnew; srow[row] = s; corr[row] = cf;
        }
        __syncthreads();

        // ---- rescale accumulators, then P @ V^T ----
#pragma unroll
        for (int r = 0; r < 8; ++r) {
            const float cf = corr[mt * 16 + r + hi8];
            oR0[r] *= cf; oI0[r] *= cf; oR1[r] *= cf; oI1[r] *= cf;
        }
#pragma unroll
        for (int ks = 0; ks < 64; ks += 32) {
            Frag p   = load_fragA(Ps, arow, 72, ks, lane);
            Frag vr0 = load_fragB(Vsr, (ntp + 0) * 16 + ln, 72, ks, lane);
            Frag vi0 = load_fragB(Vsi, (ntp + 0) * 16 + ln, 72, ks, lane);
            Frag vr1 = load_fragB(Vsr, (ntp + 1) * 16 + ln, 72, ks, lane);
            Frag vi1 = load_fragB(Vsi, (ntp + 1) * 16 + ln, 72, ks, lane);
            oR0 = wmma_bf16(p, vr0, oR0);
            oI0 = wmma_bf16(p, vi0, oI0);
            oR1 = wmma_bf16(p, vr1, oR1);
            oI1 = wmma_bf16(p, vi1, oI1);
        }
        __syncthreads();
    }

    // ---- normalize and store (B,S,F) bf16 at columns h*64.. ----
#pragma unroll
    for (int r = 0; r < 8; ++r) {
        const int row = mt * 16 + r + hi8;
        const float inv = 1.0f / srow[row];
        const size_t base = (size_t)(b * S_ + q0 + row) * F_ + h * 64;
        {
            const int col = (ntp + 0) * 16 + ln;
            Or[base + col] = f2bf(oR0[r] * inv);
            Oi[base + col] = f2bf(oI0[r] * inv);
        }
        {
            const int col = (ntp + 1) * 16 + ln;
            Or[base + col] = f2bf(oR1[r] * inv);
            Oi[base + col] = f2bf(oI1[r] * inv);
        }
    }
}

// =====================================================================
extern "C" void kernel_launch(void* const* d_in, const int* in_sizes, int n_in,
                              void* d_out, int out_size, void* d_ws, size_t ws_size,
                              hipStream_t stream)
{
    (void)in_sizes; (void)n_in; (void)out_size; (void)ws_size;
    const float* x_r   = (const float*)d_in[0];
    const float* x_i   = (const float*)d_in[1];
    const float* ctx_r = (const float*)d_in[2];
    const float* ctx_i = (const float*)d_in[3];
    const float* mask  = (const float*)d_in[4];
    const float* Wqr = (const float*)d_in[5];
    const float* Wqi = (const float*)d_in[6];
    const float* bqr = (const float*)d_in[7];
    const float* bqi = (const float*)d_in[8];
    const float* Wkr = (const float*)d_in[9];
    const float* Wki = (const float*)d_in[10];
    const float* bkr = (const float*)d_in[11];
    const float* bki = (const float*)d_in[12];
    const float* Wvr = (const float*)d_in[13];
    const float* Wvi = (const float*)d_in[14];
    const float* bvr = (const float*)d_in[15];
    const float* bvi = (const float*)d_in[16];
    const float* Wor = (const float*)d_in[17];
    const float* Woi = (const float*)d_in[18];
    const float* bor = (const float*)d_in[19];
    const float* boi = (const float*)d_in[20];

    char* ws = (char*)d_ws;
    size_t off = 0;
    auto carve = [&](size_t bytes) {
        void* p = ws + off;
        off += (bytes + 255) & ~(size_t)255;
        return p;
    };
    const size_t QB = (size_t)B_ * S_ * F_ * 2;   // bf16 bytes for (B,S,F)
    const size_t KB = (size_t)B_ * LC_ * F_ * 2;  // bf16 bytes for (B,Lc,F)
    unsigned short* Qrw  = (unsigned short*)carve(QB);
    unsigned short* Qiw  = (unsigned short*)carve(QB);
    unsigned short* Krw  = (unsigned short*)carve(KB);
    unsigned short* Kiw  = (unsigned short*)carve(KB);
    unsigned short* Vtrw = (unsigned short*)carve(KB);
    unsigned short* Vtiw = (unsigned short*)carve(KB);
    unsigned short* Orw  = (unsigned short*)carve(QB);
    unsigned short* Oiw  = (unsigned short*)carve(QB);

    dim3 blk(256);

    // Q projection: (B*S, F) x (F, F), complex
    cgemm_kernel<0, false><<<dim3(F_ / 64, (B_ * S_) / 64), blk, 0, stream>>>(
        x_r, x_i, Wqr, Wqi, bqr, bqi, Qrw, Qiw, B_ * S_, F_, F_);
    // K projection: (B*Lc, Dc) x (Dc, F), complex
    cgemm_kernel<0, false><<<dim3(F_ / 64, (B_ * LC_) / 64), blk, 0, stream>>>(
        ctx_r, ctx_i, Wkr, Wki, bkr, bki, Krw, Kiw, B_ * LC_, DC_, F_);
    // V projection, stored transposed per head (B,H,HD,Lc)
    cgemm_kernel<1, false><<<dim3(F_ / 64, (B_ * LC_) / 64), blk, 0, stream>>>(
        ctx_r, ctx_i, Wvr, Wvi, bvr, bvi, Vtrw, Vtiw, B_ * LC_, DC_, F_);
    // fused attention
    attn_kernel<<<dim3(S_ / 64, B_ * H_), blk, ATTN_SMEM, stream>>>(
        Qrw, Qiw, Krw, Kiw, Vtrw, Vtiw, mask, Orw, Oiw);
    // output projection -> interleaved complex64 in d_out
    cgemm_kernel<2, true><<<dim3(F_ / 64, (B_ * S_) / 64), blk, 0, stream>>>(
        Orw, Oiw, Wor, Woi, bor, boi, d_out, nullptr, B_ * S_, F_, F_);
}
